// ActorCriticGnnNetwork_80788334838341
// MI455X (gfx1250) — matrix operand order repaired and checked by reading, not compiled, not run on hardware
//
#include <hip/hip_runtime.h>
#include <math.h>

// ---------------------------------------------------------------------------
// Types for CDNA5 WMMA (wave32): v_wmma_f32_16x16x32_bf16
// ---------------------------------------------------------------------------
typedef __bf16 bh;
typedef __attribute__((ext_vector_type(8)))  float v8f;
typedef __attribute__((ext_vector_type(16))) bh    v16bh;
typedef __attribute__((ext_vector_type(8)))  bh    v8bh;

union U16 { v16bh v; v8bh h[2]; };

#define GB    16                 // batch items per workgroup
#define NN    20                 // agents per batch item
#define RPG   (GB * NN)          // 320 rows per group
#define DD    13                 // input dim
#define TPG   (RPG / 16)         // 20 M-tiles per group
#define NWAVE 8                  // 256 threads = 8 waves

// ---------------------------------------------------------------------------
// LDS layout (bytes). Total ~290KB < 320KB/WGP (CDNA5). Dynamic LDS.
// ---------------------------------------------------------------------------
enum : unsigned {
  OXIN  = 0,                         // 320 x 32 bf16  padded state
  OXH   = OXIN  + RPG * 32 * 2,      // 320 x 32 bf16  padded human slice
  OM1   = OXH   + RPG * 32 * 2,      // 320 x 128 bf16 m1 (100 valid)
  OM2   = OM1   + RPG * 128 * 2,     // 320 x 64 bf16  m2 (50 valid)
  OCAT  = OM2   + RPG * 64 * 2,      // 320 x 64 bf16  [e|h|r|0] concat
  OHID  = OCAT  + RPG * 64 * 2,      // 8 waves x 16x160 bf16 scratch
  OGST  = OHID  + NWAVE * 16 * 160 * 2, // 16 x 128 bf16 gstate
  OELAT = OGST  + GB * 128 * 2,      // 320 x 16 f32   e_lat
  OSCR  = OELAT + RPG * 16 * 4,      // 320 f32 attention scores
  OWGT  = OSCR  + RPG * 4,           // 16 x 20 f32 attn weights
  OSLF  = OWGT  + GB * NN * 4,       // 16 x 8 f32 self_state
  OSLB  = OSLF  + GB * 8 * 4,        // 16 x 32 bf16 self padded (node_b in)
  ORMB  = OSLB  + GB * 32 * 2,       // 16 x 16 bf16 r_emb
  OJNT  = ORMB  + GB * 16 * 2,       // 16 x 96 bf16 joint (72 valid)
  OJ2   = OJNT  + GB * 96 * 2,       // 16 x 160 bf16
  OJ3   = OJ2   + GB * 160 * 2,      // 16 x 128 bf16
  OJ4   = OJ3   + GB * 128 * 2,      // 16 x 128 bf16
  OOUT  = OJ4   + GB * 128 * 2,      // 16 x 96 f32 final pre-softmax
  LDSZ  = OOUT  + GB * 96 * 4
};

// packed-weight pool size (bf16 elems), biases appended after (f32, padded)
#define WPOOL_EL   171520
#define BPOOL_OFF  (WPOOL_EL * 2)    // byte offset of padded bias pool

// ---------------------------------------------------------------------------
// Fragment loaders per CDNA5 16-bit WMMA VGPR layouts (ISA 7.12.2)
// A 16x32: lanes 0-15 row M=lane, K 0..7 & 16..23; lanes 16-31 K 8..15 & 24..31
// B 32x16: lanes 0-15 col N=lane, K 0..7 & 16..23; lanes 16-31 K 8..15 & 24..31
// ---------------------------------------------------------------------------
__device__ __forceinline__ v16bh frag_a(const bh* rowbase, int lane) {
  int kb = (lane & 16) ? 8 : 0;
  U16 u;
  u.h[0] = *(const v8bh*)(rowbase + kb);
  u.h[1] = *(const v8bh*)(rowbase + kb + 16);
  return u.v;
}
__device__ __forceinline__ v16bh frag_b(const bh* wT, int pitch, int n0, int k0, int lane) {
  const bh* p = wT + (n0 + (lane & 15)) * pitch + k0 + ((lane & 16) ? 8 : 0);
  U16 u;
  u.h[0] = *(const v8bh*)p;
  u.h[1] = *(const v8bh*)(p + 16);
  return u.v;
}
__device__ __forceinline__ v8f wmma_bf16(v16bh a, v16bh b, v8f c) {
  return __builtin_amdgcn_wmma_f32_16x16x32_bf16(false, a, false, b, (short)0, c, false, false);
}

// C/D layout: lane N = lane&15; VGPR i -> M = i + (lane<16 ? 0 : 8)
// biasPad is zero-padded, so no bounds guards are needed in the epilogue.
__device__ __forceinline__ void st_bf(bh* dst, int pitch, int r0, int n0, v8f acc,
                                      const float* biasPad, int relu, int lane) {
  int col = n0 + (lane & 15);
  int m0  = (lane & 16) ? 8 : 0;
  float bv = biasPad[col];
#pragma unroll
  for (int i = 0; i < 8; ++i) {
    float v = acc[i] + bv;
    if (relu) v = fmaxf(v, 0.f);
    dst[(r0 + m0 + i) * pitch + col] = (bh)v;
  }
}
__device__ __forceinline__ void st_f32(float* dst, int pitch, int r0, int n0, v8f acc,
                                       const float* biasPad, int relu, int lane) {
  int col = n0 + (lane & 15);
  int m0  = (lane & 16) ? 8 : 0;
  float bv = biasPad[col];
#pragma unroll
  for (int i = 0; i < 8; ++i) {
    float v = acc[i] + bv;
    if (relu) v = fmaxf(v, 0.f);
    dst[(r0 + m0 + i) * pitch + col] = v;
  }
}

// One dense layer on one 16-row M-tile: Y = act(X @ W^T + b)
template<int KT, int NT>
__device__ __forceinline__ void layer(const bh* xrow, const bh* wT, int wpitch,
                                      const float* biasPad, int relu,
                                      bh* y, int ypitch, int yr0, int lane) {
  v16bh a[KT];
#pragma unroll
  for (int k = 0; k < KT; ++k) a[k] = frag_a(xrow + 32 * k, lane);
#pragma unroll
  for (int n = 0; n < NT; ++n) {
    v8f acc = {};
#pragma unroll
    for (int k = 0; k < KT; ++k)
      acc = wmma_bf16(a[k], frag_b(wT, wpitch, n * 16, k * 32, lane), acc);
    st_bf(y, ypitch, yr0, n * 16, acc, biasPad, relu, lane);
  }
}

// ---------------------------------------------------------------------------
struct Params {
  const float* state;
  // zero-padded biases (in workspace)
  const float *b1a, *b1b, *b2a, *b2b, *ba1, *ba2;
  const float *bna1, *bna2, *bnb1, *bnb2, *be1, *be2, *bge;
  const float *b3a, *b3b, *b3c, *b3d;
  const float *ba3raw;  // scalar bias for attention score
  const bh *W1a, *W1b, *W2a, *W2b, *Wa1, *Wa2, *Wa3;
  const bh *Wna1, *Wna2, *Wnb1, *Wnb2, *We1, *We2, *Wge;
  const bh *W3a, *W3b, *W3c, *W3d;
  float* out;
};

__global__ __launch_bounds__(256) void actor_critic_fused(Params p) {
  extern __shared__ char smem[];
  bh*    sXin  = (bh*)(smem + OXIN);
  bh*    sXh   = (bh*)(smem + OXH);
  bh*    sM1   = (bh*)(smem + OM1);
  bh*    sM2   = (bh*)(smem + OM2);
  bh*    sCat  = (bh*)(smem + OCAT);
  bh*    sHid  = (bh*)(smem + OHID);
  bh*    sGst  = (bh*)(smem + OGST);
  float* sElat = (float*)(smem + OELAT);
  float* sScr  = (float*)(smem + OSCR);
  float* sWgt  = (float*)(smem + OWGT);
  float* sSelf = (float*)(smem + OSLF);
  bh*    sSlb  = (bh*)(smem + OSLB);
  bh*    sRmb  = (bh*)(smem + ORMB);
  bh*    sJnt  = (bh*)(smem + OJNT);
  bh*    sJ2   = (bh*)(smem + OJ2);
  bh*    sJ3   = (bh*)(smem + OJ3);
  bh*    sJ4   = (bh*)(smem + OJ4);
  float* sOutB = (float*)(smem + OOUT);

  const int tid  = threadIdx.x;
  const int lane = tid & 31;
  const int wv   = tid >> 5;
  const int b0   = blockIdx.x * GB;
  const float* st = p.state + (size_t)b0 * NN * DD;

  // ---- warm L2/WGP$ with the shared packed-weight pool (global_prefetch) ----
  if (wv == 0) {
    const char* wbase = (const char*)p.W1a;
    for (int i = lane; i < 86; i += 32)
      __builtin_prefetch(wbase + (size_t)i * 4096, 0, 0);
  }

  // ---- stage 0: bulk-zero padded LDS, then stage valid elements as bf16 ----
  {
    v8bh z = {};
    for (int i = tid * 8; i < RPG * 32; i += 256 * 8) {
      *(v8bh*)(sXin + i) = z;
      *(v8bh*)(sXh + i)  = z;
    }
    for (int i = tid * 8; i < RPG * 64; i += 256 * 8) *(v8bh*)(sCat + i) = z;
    for (int i = tid * 8; i < GB * 32; i += 256 * 8)  *(v8bh*)(sSlb + i) = z;
  }
  __syncthreads();
  for (int r = tid; r < RPG; r += 256) {
    const float* row = st + r * DD;
#pragma unroll
    for (int c = 0; c < DD; ++c) sXin[r * 32 + c] = (bh)row[c];
#pragma unroll
    for (int c = 0; c < 7; ++c) sXh[r * 32 + c] = (bh)row[6 + c];
  }
  for (int i = tid; i < GB * 8; i += 256) {
    int b = i >> 3, c = i & 7;
    float v = (c < 6) ? st[b * NN * DD + c] : 0.f;  // state[:,0,:6]
    sSelf[i] = v;
    if (c < 6) sSlb[b * 32 + c] = (bh)v;
  }
  __syncthreads();

  // ---- hoisted B fragments for the tiny embedding layers (live in VGPRs) ----
  v16bh bWe1_0 = frag_b(p.We1, 32, 0, 0, lane), bWe1_1 = frag_b(p.We1, 32, 16, 0, lane);
  v16bh bWe2_0 = frag_b(p.We2, 32, 0, 0, lane);
  v16bh bWh1_0 = frag_b(p.Wna1, 32, 0, 0, lane), bWh1_1 = frag_b(p.Wna1, 32, 16, 0, lane);
  v16bh bWh2_0 = frag_b(p.Wna2, 32, 0, 0, lane);

  // ---- stage 1: row-wise MLPs (MLP1, MLP2, edge-emb, human-emb) ----
  for (int t = wv; t < TPG; t += NWAVE) {
    int r0 = t * 16;
    bh* hid = sHid + wv * (16 * 160);
    const bh* xrow  = sXin + (r0 + (lane & 15)) * 32;
    const bh* hrow  = hid + (lane & 15) * 160;
    layer<1, 10>(xrow, p.W1a, 32, p.b1a, 1, hid, 160, 0, lane);      // 13->150
    layer<5, 8>(hrow, p.W1b, 160, p.b1b, 1, sM1, 128, r0, lane);     // 150->100
    const bh* m1row = sM1 + (r0 + (lane & 15)) * 128;
    layer<4, 8>(m1row, p.W2a, 128, p.b2a, 1, hid, 160, 0, lane);     // 100->100
    layer<4, 4>(hrow, p.W2b, 128, p.b2b, 0, sM2, 64, r0, lane);      // 100->50
    {  // edge-emb 13->32->16 into concat cols 0..15
      v16bh aX = frag_a(xrow, lane);
      v8f a0 = {}, a1 = {};
      a0 = wmma_bf16(aX, bWe1_0, a0);
      a1 = wmma_bf16(aX, bWe1_1, a1);
      st_bf(hid, 160, 0, 0, a0, p.be1, 1, lane);
      st_bf(hid, 160, 0, 16, a1, p.be1, 1, lane);
      v8f a2 = {};
      a2 = wmma_bf16(frag_a(hrow, lane), bWe2_0, a2);
      st_bf(sCat, 64, r0, 0, a2, p.be2, 0, lane);
    }
    {  // human-emb 7->32->16 into concat cols 16..31
      v16bh aH = frag_a(sXh + (r0 + (lane & 15)) * 32, lane);
      v8f a0 = {}, a1 = {};
      a0 = wmma_bf16(aH, bWh1_0, a0);
      a1 = wmma_bf16(aH, bWh1_1, a1);
      st_bf(hid, 160, 0, 0, a0, p.bna1, 1, lane);
      st_bf(hid, 160, 0, 16, a1, p.bna1, 1, lane);
      v8f a2 = {};
      a2 = wmma_bf16(frag_a(hrow, lane), bWh2_0, a2);
      st_bf(sCat + 16, 64, r0, 0, a2, p.bna2, 0, lane);
    }
  }
  if (wv == 0) {  // node_b (r_emb), 16 rows -> one tile
    bh* hid = sHid;
    const bh* srow = sSlb + (lane & 15) * 32;
    const bh* hrow = hid + (lane & 15) * 160;
    layer<1, 2>(srow, p.Wnb1, 32, p.bnb1, 1, hid, 160, 0, lane);     // 6->32
    layer<1, 1>(hrow, p.Wnb2, 32, p.bnb2, 0, sRmb, 16, 0, lane);     // 32->16
  }
  __syncthreads();

  // ---- broadcast r_emb into concat cols 32..47; compute gstate = mean_n(m1) ----
  for (int i = tid; i < RPG * 16; i += 256) {
    int r = i >> 4, c = i & 15;
    sCat[r * 64 + 32 + c] = sRmb[(r / NN) * 16 + c];
  }
  for (int i = tid; i < GB * 128; i += 256) {
    int b = i >> 7, c = i & 127;
    float s = 0.f;
    for (int n = 0; n < NN; ++n) s += (float)sM1[(b * NN + n) * 128 + c];
    sGst[i] = (bh)(s * (1.f / NN));
  }
  __syncthreads();

  // ---- stage 2: attention MLP (200->100->100->1) + GNN edge (48->16) ----
  v16bh bWge0 = frag_b(p.Wge, 64, 0, 0, lane);
  v16bh bWge1 = frag_b(p.Wge, 64, 0, 32, lane);
  for (int t = wv; t < TPG; t += NWAVE) {
    int r0 = t * 16;
    bh* hid = sHid + wv * (16 * 160);
    const bh* hrow = hid + (lane & 15) * 160;
    int myrow = r0 + (lane & 15);
    const bh* m1row = sM1 + myrow * 128;
    const bh* grow  = sGst + (myrow / NN) * 128;
    v16bh af[8];
#pragma unroll
    for (int k = 0; k < 4; ++k) af[k] = frag_a(m1row + 32 * k, lane);
#pragma unroll
    for (int k = 0; k < 4; ++k) af[4 + k] = frag_a(grow + 32 * k, lane);
#pragma unroll
    for (int n = 0; n < 8; ++n) {
      v8f acc = {};
#pragma unroll
      for (int k = 0; k < 8; ++k)
        acc = wmma_bf16(af[k], frag_b(p.Wa1, 256, n * 16, k * 32, lane), acc);
      st_bf(hid, 160, 0, n * 16, acc, p.ba1, 1, lane);
    }
    layer<4, 8>(hrow, p.Wa2, 128, p.ba2, 1, hid, 160, 0, lane);
    {
      v8f acc = {};
#pragma unroll
      for (int k = 0; k < 4; ++k)
        acc = wmma_bf16(frag_a(hrow + 32 * k, lane), frag_b(p.Wa3, 128, 0, k * 32, lane), acc);
      if ((lane & 15) == 0) {  // only column 0 holds the score
        int m0 = (lane & 16) ? 8 : 0;
        float bv = p.ba3raw[0];
#pragma unroll
        for (int i = 0; i < 8; ++i) sScr[r0 + m0 + i] = acc[i] + bv;
      }
    }
    {  // gnn edge: relu(cat48 @ Wge + b), K padded to 64
      const bh* crow = sCat + (r0 + (lane & 15)) * 64;
      v8f acc = {};
      acc = wmma_bf16(frag_a(crow, lane),      bWge0, acc);
      acc = wmma_bf16(frag_a(crow + 32, lane), bWge1, acc);
      st_f32(sElat, 16, r0, 0, acc, p.bge, 1, lane);
    }
  }
  __syncthreads();

  // ---- attention weights: exp(s)*(s!=0) normalized over N ----
  if (tid < GB) {
    float ssum = 0.f, w[NN];
    for (int n = 0; n < NN; ++n) {
      float sc = sScr[tid * NN + n];
      float e  = (sc != 0.f) ? __expf(sc) : 0.f;
      w[n] = e; ssum += e;
    }
    float inv = 1.f / ssum;
    for (int n = 0; n < NN; ++n) sWgt[tid * NN + n] = w[n] * inv;
  }
  __syncthreads();

  // ---- joint = [20*self | sum_n w*m2 | sum_n e_lat | 0pad] ----
  for (int i = tid; i < GB * 96; i += 256) {
    int b = i / 96, c = i % 96;
    float v = 0.f;
    if (c < 6) {
      v = (float)NN * sSelf[b * 8 + c];
    } else if (c < 56) {
      int cc = c - 6; float s = 0.f;
      for (int n = 0; n < NN; ++n)
        s += sWgt[b * NN + n] * (float)sM2[(b * NN + n) * 64 + cc];
      v = s;
    } else if (c < 72) {
      int cc = c - 56; float s = 0.f;
      for (int n = 0; n < NN; ++n) s += sElat[(b * NN + n) * 16 + cc];
      v = s;
    }
    sJnt[i] = (bh)v;
  }
  __syncthreads();

  // ---- MLP3: 72->150->100->100->81, one 16-row tile, N-tiles across waves ----
  {
    const bh* jrow = sJnt + (lane & 15) * 96;
    v16bh a0 = frag_a(jrow, lane), a1 = frag_a(jrow + 32, lane), a2 = frag_a(jrow + 64, lane);
    for (int n = wv; n < 10; n += NWAVE) {
      v8f acc = {};
      acc = wmma_bf16(a0, frag_b(p.W3a, 96, n * 16, 0, lane), acc);
      acc = wmma_bf16(a1, frag_b(p.W3a, 96, n * 16, 32, lane), acc);
      acc = wmma_bf16(a2, frag_b(p.W3a, 96, n * 16, 64, lane), acc);
      st_bf(sJ2, 160, 0, n * 16, acc, p.b3a, 1, lane);
    }
  }
  __syncthreads();
  {
    const bh* jrow = sJ2 + (lane & 15) * 160;
    v16bh a[5];
#pragma unroll
    for (int k = 0; k < 5; ++k) a[k] = frag_a(jrow + 32 * k, lane);
    v8f acc = {};
#pragma unroll
    for (int k = 0; k < 5; ++k)
      acc = wmma_bf16(a[k], frag_b(p.W3b, 160, wv * 16, k * 32, lane), acc);
    st_bf(sJ3, 128, 0, wv * 16, acc, p.b3b, 1, lane);
  }
  __syncthreads();
  {
    const bh* jrow = sJ3 + (lane & 15) * 128;
    v16bh a[4];
#pragma unroll
    for (int k = 0; k < 4; ++k) a[k] = frag_a(jrow + 32 * k, lane);
    v8f acc = {};
#pragma unroll
    for (int k = 0; k < 4; ++k)
      acc = wmma_bf16(a[k], frag_b(p.W3c, 128, wv * 16, k * 32, lane), acc);
    st_bf(sJ4, 128, 0, wv * 16, acc, p.b3c, 1, lane);
  }
  __syncthreads();
  if (wv < 6) {
    const bh* jrow = sJ4 + (lane & 15) * 128;
    v16bh a[4];
#pragma unroll
    for (int k = 0; k < 4; ++k) a[k] = frag_a(jrow + 32 * k, lane);
    v8f acc = {};
#pragma unroll
    for (int k = 0; k < 4; ++k)
      acc = wmma_bf16(a[k], frag_b(p.W3d, 128, wv * 16, k * 32, lane), acc);
    st_f32(sOutB, 96, 0, wv * 16, acc, p.b3d, 0, lane);
  }
  __syncthreads();

  // ---- output: value + softmax over cols 1..80 ----
  if (tid < GB) {
    const float* o = sOutB + tid * 96;
    float* g = p.out + (size_t)(b0 + tid) * 81;
    float m = -1e30f;
    for (int c = 1; c < 81; ++c) m = fmaxf(m, o[c]);
    float s = 0.f;
    for (int c = 1; c < 81; ++c) s += __expf(o[c] - m);
    float inv = 1.f / s;
    g[0] = o[0];
    for (int c = 1; c < 81; ++c) g[c] = __expf(o[c] - m) * inv;
  }
}

// ---------------------------------------------------------------------------
// Weight pre-pack: f32 (in x out) -> bf16 transposed (outPad x inPad), zero
// padded; bias -> f32 zero-padded to outPad. blkRows/blkPad support the
// attention concat (two 100-blocks each padded to 128).
// ---------------------------------------------------------------------------
__global__ void prep_weight(const float* W, const float* bsrc, bh* dst, float* bdst,
                            int in_dim, int out_dim, int inPad, int outPad,
                            int blkRows, int blkPad) {
  int o = blockIdx.x;
  int k = threadIdx.x;
  if (o >= outPad || k >= inPad) return;
  if (k == 0) bdst[o] = (o < out_dim) ? bsrc[o] : 0.f;
  float v = 0.f;
  if (o < out_dim) {
    int blk = k / blkPad;
    int kk  = k - blk * blkPad;
    int i   = blk * blkRows + kk;
    if (kk < blkRows && i < in_dim) v = W[(size_t)i * out_dim + o];
  }
  dst[(size_t)o * inPad + k] = (bh)v;
}

// ---------------------------------------------------------------------------
extern "C" void kernel_launch(void* const* d_in, const int* in_sizes, int n_in,
                              void* d_out, int out_size, void* d_ws, size_t ws_size,
                              hipStream_t stream) {
  (void)in_sizes; (void)n_in; (void)out_size; (void)ws_size;
  bh*    ws  = (bh*)d_ws;
  float* wsb = (float*)((char*)d_ws + BPOOL_OFF);

  struct WD { int src, in, out, inPad, outPad, blkRows, blkPad; size_t off, boff; };
  static const WD wd[18] = {
    { 1,  13, 150,  32, 160,  13,  32,      0,    0},  // W1a
    { 3, 150, 100, 160, 128, 150, 160,   5120,  160},  // W1b
    { 5, 100, 100, 128, 128, 100, 128,  25600,  288},  // W2a
    { 7, 100,  50, 128,  64, 100, 128,  41984,  416},  // W2b
    { 9, 200, 100, 256, 128, 100, 128,  50176,  480},  // Wa1 (split 100|100)
    {11, 100, 100, 128, 128, 100, 128,  82944,  608},  // Wa2
    {13, 100,   1, 128,  16, 100, 128,  99328,  736},  // Wa3
    {15,   7,  32,  32,  32,   7,  32, 101376,  752},  // Wna1
    {17,  32,  16,  32,  16,  32,  32, 102400,  784},  // Wna2
    {19,   6,  32,  32,  32,   6,  32, 102912,  800},  // Wnb1
    {21,  32,  16,  32,  16,  32,  32, 103936,  816},  // Wnb2
    {23,  13,  32,  32,  32,  13,  32, 104448,  832},  // We1
    {25,  32,  16,  32,  16,  32,  32, 105472,  864},  // We2
    {27,  48,  16,  64,  16,  48,  64, 105984,  880},  // Wge
    {33,  72, 150,  96, 160,  72,  96, 107008,  896},  // W3a
    {35, 150, 100, 160, 128, 150, 160, 122368, 1056},  // W3b
    {37, 100, 100, 128, 128, 100, 128, 142848, 1184},  // W3c
    {39, 100,  81, 128,  96, 100, 128, 159232, 1312},  // W3d
  };
  for (int i = 0; i < 18; ++i) {
    prep_weight<<<dim3(wd[i].outPad), dim3(wd[i].inPad), 0, stream>>>(
        (const float*)d_in[wd[i].src], (const float*)d_in[wd[i].src + 1],
        ws + wd[i].off, wsb + wd[i].boff,
        wd[i].in, wd[i].out, wd[i].inPad, wd[i].outPad, wd[i].blkRows, wd[i].blkPad);
  }

  Params p;
  p.state = (const float*)d_in[0];
  p.b1a  = wsb + 0;    p.b1b  = wsb + 160;
  p.b2a  = wsb + 288;  p.b2b  = wsb + 416;
  p.ba1  = wsb + 480;  p.ba2  = wsb + 608;
  p.bna1 = wsb + 752;  p.bna2 = wsb + 784;
  p.bnb1 = wsb + 800;  p.bnb2 = wsb + 816;
  p.be1  = wsb + 832;  p.be2  = wsb + 864;
  p.bge  = wsb + 880;
  p.b3a  = wsb + 896;  p.b3b  = wsb + 1056;
  p.b3c  = wsb + 1184; p.b3d  = wsb + 1312;
  p.ba3raw = (const float*)d_in[14];
  p.W1a = ws + 0;      p.W1b = ws + 5120;   p.W2a = ws + 25600;  p.W2b = ws + 41984;
  p.Wa1 = ws + 50176;  p.Wa2 = ws + 82944;  p.Wa3 = ws + 99328;
  p.Wna1 = ws + 101376; p.Wna2 = ws + 102400;
  p.Wnb1 = ws + 102912; p.Wnb2 = ws + 103936;
  p.We1 = ws + 104448; p.We2 = ws + 105472; p.Wge = ws + 105984;
  p.W3a = ws + 107008; p.W3b = ws + 122368; p.W3c = ws + 142848; p.W3d = ws + 159232;
  p.out = (float*)d_out;

  actor_critic_fused<<<dim3(8192 / GB), dim3(256), LDSZ, stream>>>(p);
}
